// MultiHeadAttention_35536559407376
// MI455X (gfx1250) — matrix operand order repaired and checked by reading, not compiled, not run on hardware
//
#include <hip/hip_runtime.h>
#include <hip/hip_bf16.h>

// MI455X MHA forward. bf16 WMMA (v_wmma_f32_16x16x32_bf16) for all GEMMs,
// fused energy+softmax holding a 32x2048 energy strip in LDS (~265KB of the
// 320KB/WGP) so the 1.07GB attn matrix is written once and read once.
// Tensor Data Mover (tensor_load_to_lds) stages all pure-bf16 tiles
// (weights, V tiles, Q strips) asynchronously with double buffering;
// TDM hardware padding (4 DW every 16 DW) realizes the 80B LDS row stride.
//
// Workspace (elements are bf16/ushort):
//   [0,4M)      Wq,Wk,Wv,Wo bf16
//   [4M,12M)    Q bf16  [B,H,S,Dh]     (reused as ctx [B,S,E] later)
//   [12M,20M)   K bf16  [B,H,S,Dh]
//   [20M,28M)   V^T bf16 [B,H,Dh,S]

#define EMBED 1024
#define HEADS 16
#define HD    64
#define BATCH 4
#define SEQ   2048
#define MTOT  (BATCH * SEQ)      // 8192
#define QB    32                 // q-rows per fused softmax workgroup
#define ESTR  2068               // energy LDS row stride in floats

typedef __attribute__((ext_vector_type(16))) __bf16 v16bf;
typedef __attribute__((ext_vector_type(8)))  float  v8f;

union FragU { uint4 u[2]; unsigned short s[16]; v16bf v; };
union AccU  { v8f v; float f[8]; };

__device__ __forceinline__ unsigned short f2bf(float x) {
  union { float f; unsigned u; } c; c.f = x;
  unsigned r = c.u + 0x7FFFu + ((c.u >> 16) & 1u);   // round-to-nearest-even
  return (unsigned short)(r >> 16);
}
__device__ __forceinline__ unsigned pk2(float a, float b) {
  return (unsigned)f2bf(a) | ((unsigned)f2bf(b) << 16);
}

// ---------------------------------------------------------------------------
// Tensor Data Mover: 2D bf16 tile load, global -> LDS.
// This toolchain exposes the 6-arg builtin:
//   (uint32x4 g0, int32x8 g1, int32x4 g2, int32x4 g3, int32x8 pad, i32 cpol)
// ---------------------------------------------------------------------------
#if defined(__has_builtin)
#if __has_builtin(__builtin_amdgcn_tensor_load_to_lds) && \
    __has_builtin(__builtin_amdgcn_s_wait_tensorcnt)
#define HAVE_TDM 1
#endif
#endif
#ifndef HAVE_TDM
#define HAVE_TDM 0
#endif

#if HAVE_TDM
typedef unsigned tdm_v4u __attribute__((ext_vector_type(4)));
typedef int      tdm_v8i __attribute__((ext_vector_type(8)));
typedef int      tdm_v4i __attribute__((ext_vector_type(4)));

// tile_w bf16 elements per row, tile_h rows, global row stride = `stride`
// elements. pad80: insert 4 DWORDs of LDS padding every 16 DWORDs (64B->80B).
__device__ __forceinline__ void tdm_load_tile_bf16(
    unsigned lds_byte_addr, const unsigned short* g, unsigned tile_w,
    unsigned tile_h, unsigned stride, bool pad80) {
  const unsigned long long ga = (unsigned long long)(size_t)g;
  tdm_v4u g0;
  g0.x = 1u;                                            // count=1, user D#
  g0.y = lds_byte_addr;                                 // lds_addr
  g0.z = (unsigned)ga;                                  // global_addr lo
  g0.w = (unsigned)((ga >> 32) & 0x1FFFFFFu) | (2u << 30);  // addr hi | type=2
  unsigned w0 = (1u << 16);                             // data_size = 2 bytes
  if (pad80) w0 |= (1u << 20) | (3u << 22) | (3u << 25);  // pad 4DW per 16DW
  const unsigned td0 = stride;        // tensor_dim0: stride-sized -> no OOB
  const unsigned td1 = 0x400000u;     // tensor_dim1: large -> no OOB
  tdm_v8i g1;
  g1[0] = (int)w0;
  g1[1] = (int)((td0 & 0xFFFFu) << 16);                 // dim0[15:0]
  g1[2] = (int)((td0 >> 16) | ((td1 & 0xFFFFu) << 16)); // dim0[31:16],dim1 lo
  g1[3] = (int)((td1 >> 16) | (tile_w << 16));          // dim1 hi, tile_dim0
  g1[4] = (int)(tile_h & 0xFFFFu);                      // tile_dim1
  g1[5] = (int)stride;                                  // dim0_stride lo32
  g1[6] = 0;
  g1[7] = 0;
  tdm_v4i z4 = {0, 0, 0, 0};
  tdm_v8i z8 = {0, 0, 0, 0, 0, 0, 0, 0};
  __builtin_amdgcn_tensor_load_to_lds(g0, g1, z4, z4, z8, 0);
}
__device__ __forceinline__ unsigned lds_addr_of(const void* p) {
  return (unsigned)(size_t)p;   // low 32 bits of generic ptr = LDS offset
}
#endif

// ---------------------------------------------------------------------------
// K0: fp32 -> bf16 weight conversion (once; weights re-read by 64 M-blocks)
// ---------------------------------------------------------------------------
__global__ __launch_bounds__(256) void wconv_kernel(const float* __restrict__ w,
                                                    unsigned short* __restrict__ o) {
  int i = (blockIdx.x * 256 + threadIdx.x) * 4;
  float4 f = *(const float4*)(w + i);
  uint2 p;
  p.x = pk2(f.x, f.y);
  p.y = pk2(f.z, f.w);
  *(uint2*)(o + i) = p;
}

// ---------------------------------------------------------------------------
// K1/K5: tiled WMMA GEMM  C[m,n] = X[m,:] . W[n,:] + bias[n]
//   128x128 block, BK=32, 8 waves (4x2), wave = 2x4 16x16 accumulators.
//   MODE 0: X fp32 -> Q bf16 [B,H,S,Dh]
//   MODE 1: X fp32 -> K bf16 [B,H,S,Dh]
//   MODE 2: X fp32 -> V^T bf16 [B,H,Dh,S]
//   MODE 3: X bf16 (ctx) -> fp32 out [B,S,E]
//   Weight tiles stream via TDM, double-buffered (copy kb+1 || compute kb).
// ---------------------------------------------------------------------------
template <int MODE>
__global__ __launch_bounds__(256) void gemm_proj_kernel(
    const void* __restrict__ Xv, const unsigned short* __restrict__ Wb,
    const float* __restrict__ bias, void* __restrict__ dst) {
  __shared__ __align__(16) unsigned short As[128 * 40];
  __shared__ __align__(16) unsigned short Bs[2][128 * 40];

  const int tid  = threadIdx.x;
  const int wave = tid >> 5;
  const int lane = tid & 31;
  const int l15  = lane & 15;
  const int half = lane >> 4;
  const int wm   = wave >> 1;
  const int wn   = wave & 1;
  const int rowX0 = blockIdx.x * 128;
  const int rowW0 = blockIdx.y * 128;
  const int KIT = EMBED / 32;

  AccU acc[2][4] = {};

  const int sr = tid >> 1;
  const int sh = tid & 1;

#if HAVE_TDM
  const unsigned bsAddr0 = lds_addr_of(&Bs[0][0]);
  const unsigned bsAddr1 = lds_addr_of(&Bs[1][0]);
  if (wave == 0)
    tdm_load_tile_bf16(bsAddr0, Wb + (size_t)rowW0 * EMBED, 32, 128, EMBED, true);
#endif

  for (int kb = 0; kb < KIT; ++kb) {
    __syncthreads();   // previous iteration's fragment reads are done

    // ---- stage A tile (128x32) ----
    if constexpr (MODE < 3) {
      const float* X = (const float*)Xv;
      const float4* sp =
          (const float4*)(X + (size_t)(rowX0 + sr) * EMBED + kb * 32 + sh * 16);
      float4 f0 = sp[0], f1 = sp[1], f2 = sp[2], f3 = sp[3];
      uint4 p0, p1;
      p0.x = pk2(f0.x, f0.y); p0.y = pk2(f0.z, f0.w);
      p0.z = pk2(f1.x, f1.y); p0.w = pk2(f1.z, f1.w);
      p1.x = pk2(f2.x, f2.y); p1.y = pk2(f2.z, f2.w);
      p1.z = pk2(f3.x, f3.y); p1.w = pk2(f3.z, f3.w);
      uint4* dp = (uint4*)(As + sr * 40 + sh * 16);
      dp[0] = p0; dp[1] = p1;
    }
#if !HAVE_TDM
    if constexpr (MODE == 3) {
      const unsigned short* X = (const unsigned short*)Xv;
      const uint4* sp =
          (const uint4*)(X + (size_t)(rowX0 + sr) * EMBED + kb * 32 + sh * 16);
      uint4* dp = (uint4*)(As + sr * 40 + sh * 16);
      dp[0] = sp[0]; dp[1] = sp[1];
    }
    {  // manual B stage
      const uint4* sp =
          (const uint4*)(Wb + (size_t)(rowW0 + sr) * EMBED + kb * 32 + sh * 16);
      uint4* dp = (uint4*)(&Bs[kb & 1][0] + sr * 40 + sh * 16);
      dp[0] = sp[0]; dp[1] = sp[1];
    }
#else
    if (wave == 0) {
      if constexpr (MODE == 3)
        tdm_load_tile_bf16(lds_addr_of(As),
                           (const unsigned short*)Xv + (size_t)rowX0 * EMBED + kb * 32,
                           32, 128, EMBED, true);
      if (kb + 1 < KIT) {
        tdm_load_tile_bf16(((kb + 1) & 1) ? bsAddr1 : bsAddr0,
                           Wb + (size_t)rowW0 * EMBED + (kb + 1) * 32,
                           32, 128, EMBED, true);
        __builtin_amdgcn_s_wait_tensorcnt(1);   // kb's tiles have landed
      } else {
        __builtin_amdgcn_s_wait_tensorcnt(0);
      }
    }
#endif
    __syncthreads();

    const unsigned short* Bsb = &Bs[kb & 1][0];
    FragU a[2], b[4];
#pragma unroll
    for (int i = 0; i < 2; ++i) {
      const int m = wm * 32 + i * 16 + l15;
      const uint4* p = (const uint4*)(As + m * 40 + half * 16);
      a[i].u[0] = p[0];
      a[i].u[1] = p[1];
    }
#pragma unroll
    for (int j = 0; j < 4; ++j) {
      const int n = wn * 64 + j * 16 + l15;
      const uint4* p = (const uint4*)(Bsb + n * 40 + half * 16);
      b[j].u[0] = p[0];
      b[j].u[1] = p[1];
    }
#pragma unroll
    for (int i = 0; i < 2; ++i)
#pragma unroll
      for (int j = 0; j < 4; ++j)
        acc[i][j].v = __builtin_amdgcn_wmma_f32_16x16x32_bf16(
            false, a[i].v, false, b[j].v, (short)0, acc[i][j].v, false, false);
  }

  // ---- epilogue ----
#pragma unroll
  for (int i = 0; i < 2; ++i) {
    const int m0 = rowX0 + wm * 32 + i * 16 + half * 8;
#pragma unroll
    for (int j = 0; j < 4; ++j) {
      const int n = rowW0 + wn * 64 + j * 16 + l15;
      const float bv = bias[n];
      if constexpr (MODE == 3) {
        float* out = (float*)dst;
#pragma unroll
        for (int r = 0; r < 8; ++r)
          out[(size_t)(m0 + r) * EMBED + n] = acc[i][j].f[r] + bv;
      } else if constexpr (MODE == 2) {
        unsigned short* out = (unsigned short*)dst;
        const int bI = m0 >> 11, s = m0 & (SEQ - 1);
        const int h = n >> 6, d = n & (HD - 1);
        union { unsigned short s16[8]; uint4 u; } pk;
#pragma unroll
        for (int r = 0; r < 8; ++r) pk.s16[r] = f2bf(acc[i][j].f[r] + bv);
        *(uint4*)(out + ((size_t)(bI * HEADS + h) * HD + d) * SEQ + s) = pk.u;
      } else {
        unsigned short* out = (unsigned short*)dst;
        const int h = n >> 6, d = n & (HD - 1);
#pragma unroll
        for (int r = 0; r < 8; ++r) {
          const int mg = m0 + r;
          const int bI = mg >> 11, s = mg & (SEQ - 1);
          out[((size_t)(bI * HEADS + h) * SEQ + s) * HD + d] =
              f2bf(acc[i][j].f[r] + bv);
        }
      }
    }
  }
}

// ---------------------------------------------------------------------------
// K2: fused energy + softmax. WG owns 32 q-rows x 2048 k-cols of one (b,h).
// Q strip staged once via TDM (was loaded 8x redundantly); energy strip in
// LDS; attn written to HBM exactly once.
// ---------------------------------------------------------------------------
__global__ __launch_bounds__(256) void attn_softmax_kernel(
    const unsigned short* __restrict__ Qh, const unsigned short* __restrict__ Kh,
    float* __restrict__ attn) {
  extern __shared__ float eLDS[];                        // [QB][ESTR]
  __shared__ float red[QB][9];
  __shared__ __align__(16) unsigned short Qs[32 * 80];   // 32 rows, padded 160B

  const int tid  = threadIdx.x;
  const int wave = tid >> 5;
  const int lane = tid & 31;
  const int l15  = lane & 15;
  const int half = lane >> 4;
  const int qblk = blockIdx.x;
  const int bh   = blockIdx.y;

  const unsigned short* Qp = Qh + (size_t)bh * SEQ * HD + (size_t)qblk * QB * HD;
  const unsigned short* Kp = Kh + (size_t)bh * SEQ * HD;

  // ---- stage Q strip (32x64 bf16) into LDS once ----
#if HAVE_TDM
  if (wave == 0) {
    tdm_load_tile_bf16(lds_addr_of(Qs), Qp, HD, QB, HD, true);
    __builtin_amdgcn_s_wait_tensorcnt(0);
  }
#else
  {
    const int r = tid >> 3, e = (tid & 7) * 8;   // 16B per thread
    const uint4* sp = (const uint4*)(Qp + (size_t)r * HD + e);
    *(uint4*)(Qs + r * 80 + (e >> 5) * 40 + (e & 31)) = sp[0];
  }
#endif
  __syncthreads();

  // ---- phase A: energy via WMMA into LDS, scaled by 1/sqrt(64) ----
  FragU a[2][2];
#pragma unroll
  for (int mt = 0; mt < 2; ++mt)
#pragma unroll
    for (int ks = 0; ks < 2; ++ks) {
      const uint4* p =
          (const uint4*)(Qs + (mt * 16 + l15) * 80 + ks * 40 + half * 16);
      a[mt][ks].u[0] = p[0];
      a[mt][ks].u[1] = p[1];
    }

  const int colbase = wave * 256;
  for (int nt = 0; nt < 16; ++nt) {
    const int ncol = colbase + nt * 16;
    AccU acc0 = {}, acc1 = {};
#pragma unroll
    for (int ks = 0; ks < 2; ++ks) {
      FragU bf;
      const uint4* p =
          (const uint4*)(Kp + (size_t)(ncol + l15) * HD + ks * 32 + half * 16);
      bf.u[0] = p[0];
      bf.u[1] = p[1];
      acc0.v = __builtin_amdgcn_wmma_f32_16x16x32_bf16(
          false, a[0][ks].v, false, bf.v, (short)0, acc0.v, false, false);
      acc1.v = __builtin_amdgcn_wmma_f32_16x16x32_bf16(
          false, a[1][ks].v, false, bf.v, (short)0, acc1.v, false, false);
    }
#pragma unroll
    for (int r = 0; r < 8; ++r) {
      eLDS[(size_t)(half * 8 + r) * ESTR + ncol + l15]      = acc0.f[r] * 0.125f;
      eLDS[(size_t)(16 + half * 8 + r) * ESTR + ncol + l15] = acc1.f[r] * 0.125f;
    }
  }
  __syncthreads();

  // ---- phase B: softmax on the LDS strip, single HBM write ----
  const int row = tid >> 3;
  const int ch  = tid & 7;
  float* er = eLDS + (size_t)row * ESTR;

  float mx = -3.4e38f;
  for (int i = 0; i < 64; ++i) {
    float4 v = *(float4*)(er + (i * 8 + ch) * 4);
    mx = fmaxf(mx, fmaxf(fmaxf(v.x, v.y), fmaxf(v.z, v.w)));
  }
  red[row][ch] = mx;
  __syncthreads();
  float rmax = red[row][0];
#pragma unroll
  for (int k = 1; k < 8; ++k) rmax = fmaxf(rmax, red[row][k]);
  __syncthreads();

  float sum = 0.f;
  for (int i = 0; i < 64; ++i) {
    float4* p = (float4*)(er + (i * 8 + ch) * 4);
    float4 v = *p;
    v.x = __expf(v.x - rmax); v.y = __expf(v.y - rmax);
    v.z = __expf(v.z - rmax); v.w = __expf(v.w - rmax);
    sum += (v.x + v.y) + (v.z + v.w);
    *p = v;
  }
  red[row][ch] = sum;
  __syncthreads();
  float rs = red[row][0];
#pragma unroll
  for (int k = 1; k < 8; ++k) rs += red[row][k];
  const float inv = 1.0f / rs;

  float* orow = attn + ((size_t)bh * SEQ + qblk * QB + row) * SEQ;
  for (int i = 0; i < 64; ++i) {
    float4 v = *(float4*)(er + (i * 8 + ch) * 4);
    v.x *= inv; v.y *= inv; v.z *= inv; v.w *= inv;
    *(float4*)(orow + (i * 8 + ch) * 4) = v;
  }
}

// ---------------------------------------------------------------------------
// K4: ctx = attn @ V. WG = 128 q-rows of one (b,h). V tiles (shared by all
// 8 waves) stream via TDM, double-buffered; attn fp32 -> bf16 A-frags read
// one full 128B line per lane pair per k-step, with prefetch 4 steps ahead.
// ---------------------------------------------------------------------------
__global__ __launch_bounds__(256) void attn_v_kernel(
    const float* __restrict__ attn, const unsigned short* __restrict__ Vt,
    unsigned short* __restrict__ ctx) {
  __shared__ __align__(16) unsigned short Vs[2][64 * 40];

  const int tid  = threadIdx.x;
  const int wave = tid >> 5;
  const int lane = tid & 31;
  const int l15  = lane & 15;
  const int half = lane >> 4;
  const int bh   = blockIdx.y;
  const float* Ap = attn + (size_t)bh * SEQ * SEQ;
  const unsigned short* Vp = Vt + (size_t)bh * HD * SEQ;
  const int row0 = blockIdx.x * 128 + wave * 16;
  const float* rpBase = Ap + (size_t)(row0 + l15) * SEQ;

  AccU acc[4] = {};

#if HAVE_TDM
  const unsigned vsAddr0 = lds_addr_of(&Vs[0][0]);
  const unsigned vsAddr1 = lds_addr_of(&Vs[1][0]);
  if (wave == 0)
    tdm_load_tile_bf16(vsAddr0, Vp, 32, HD, SEQ, true);
#endif

  for (int kb = 0; kb < SEQ / 32; ++kb) {
    __syncthreads();   // readers of the buffer about to be overwritten are done
#if HAVE_TDM
    if (wave == 0) {
      if (kb + 1 < SEQ / 32) {
        tdm_load_tile_bf16(((kb + 1) & 1) ? vsAddr1 : vsAddr0,
                           Vp + (size_t)(kb + 1) * 32, 32, HD, SEQ, true);
        __builtin_amdgcn_s_wait_tensorcnt(1);
      } else {
        __builtin_amdgcn_s_wait_tensorcnt(0);
      }
    }
#else
    {
      const int vr = tid >> 2, vp4 = (tid & 3) * 8;
      const uint4* sp = (const uint4*)(Vp + (size_t)vr * SEQ + kb * 32 + vp4);
      *(uint4*)(&Vs[kb & 1][0] + vr * 40 + vp4) = sp[0];
    }
#endif

    // A fragment from attn (global fp32 -> bf16)
    FragU a;
    {
      const float* rp = rpBase + kb * 32;
      if (kb + 4 < SEQ / 32) __builtin_prefetch(rp + 4 * 32, 0, 0);
#pragma unroll
      for (int v = 0; v < 8; ++v) {
        const int k0 = half * 16 + v * 2;
        float2 f = *(const float2*)(rp + k0);
        a.s[2 * v]     = f2bf(f.x);
        a.s[2 * v + 1] = f2bf(f.y);
      }
    }
    __syncthreads();   // V tile kb visible to all waves

    const unsigned short* Vsb = &Vs[kb & 1][0];
#pragma unroll
    for (int nt = 0; nt < 4; ++nt) {
      FragU bf;
      const uint4* p = (const uint4*)(Vsb + (nt * 16 + l15) * 40 + half * 16);
      bf.u[0] = p[0];
      bf.u[1] = p[1];
      acc[nt].v = __builtin_amdgcn_wmma_f32_16x16x32_bf16(
          false, a.v, false, bf.v, (short)0, acc[nt].v, false, false);
    }
  }

  const int b = bh >> 4, h = bh & 15;
#pragma unroll
  for (int nt = 0; nt < 4; ++nt) {
    const int col = h * HD + nt * 16 + l15;
#pragma unroll
    for (int r = 0; r < 8; ++r) {
      const int s = row0 + half * 8 + r;
      ctx[(size_t)(b * SEQ + s) * EMBED + col] = f2bf(acc[nt].f[r]);
    }
  }
}

// ---------------------------------------------------------------------------
extern "C" void kernel_launch(void* const* d_in, const int* in_sizes, int n_in,
                              void* d_out, int out_size, void* d_ws, size_t ws_size,
                              hipStream_t stream) {
  (void)in_sizes; (void)n_in; (void)out_size; (void)ws_size;
  const float* key   = (const float*)d_in[0];
  const float* query = (const float*)d_in[1];
  const float* value = (const float*)d_in[2];
  const float* Wk = (const float*)d_in[3];
  const float* bk = (const float*)d_in[4];
  const float* Wq = (const float*)d_in[5];
  const float* bq = (const float*)d_in[6];
  const float* Wv = (const float*)d_in[7];
  const float* bv = (const float*)d_in[8];
  const float* Wo = (const float*)d_in[9];
  const float* bo = (const float*)d_in[10];

  float* out  = (float*)d_out;
  float* attn = out + (size_t)MTOT * EMBED;

  const size_t WSZ = (size_t)EMBED * EMBED;
  const size_t QSZ = (size_t)MTOT * EMBED;
  unsigned short* wsW  = (unsigned short*)d_ws;
  unsigned short* wsQ  = wsW + 4 * WSZ;
  unsigned short* wsK  = wsQ + QSZ;
  unsigned short* wsVT = wsK + QSZ;
  unsigned short* ctx  = wsQ;   // safe reuse: Q consumed before attn_v runs

  const dim3 gW(EMBED * EMBED / 1024);
  wconv_kernel<<<gW, 256, 0, stream>>>(Wq, wsW + 0 * WSZ);
  wconv_kernel<<<gW, 256, 0, stream>>>(Wk, wsW + 1 * WSZ);
  wconv_kernel<<<gW, 256, 0, stream>>>(Wv, wsW + 2 * WSZ);
  wconv_kernel<<<gW, 256, 0, stream>>>(Wo, wsW + 3 * WSZ);

  const dim3 gP(MTOT / 128, EMBED / 128);
  gemm_proj_kernel<0><<<gP, 256, 0, stream>>>(query, wsW + 0 * WSZ, bq, wsQ);
  gemm_proj_kernel<1><<<gP, 256, 0, stream>>>(key,   wsW + 1 * WSZ, bk, wsK);
  gemm_proj_kernel<2><<<gP, 256, 0, stream>>>(value, wsW + 2 * WSZ, bv, wsVT);

  attn_softmax_kernel<<<dim3(SEQ / QB, BATCH * HEADS), 256,
                        (size_t)QB * ESTR * sizeof(float), stream>>>(wsQ, wsK, attn);

  attn_v_kernel<<<dim3(SEQ / 128, BATCH * HEADS), 256, 0, stream>>>(attn, wsVT, ctx);

  gemm_proj_kernel<3><<<gP, 256, 0, stream>>>(ctx, wsW + 3 * WSZ, bo, out);
}
